// GNNLayer_42588895707648
// MI455X (gfx1250) — matrix operand order repaired and checked by reading, not compiled, not run on hardware
//
#include <hip/hip_runtime.h>
#include <stdint.h>

// ---------------- constants from the reference ----------------
#define NH    4
#define NC    64
#define NHC   256
#define NU_N  40000
#define NT_N  40000
#define E1_N  250000
#define E2_N  250000
#define E3_N  250000
#define ND    256
#define NDE   32

typedef __bf16 bf16;
typedef __attribute__((ext_vector_type(16))) __bf16 v16bf;
typedef __attribute__((ext_vector_type(8)))  __bf16 v8bf;
typedef __attribute__((ext_vector_type(8)))  float  v8f;

// ---------------- helpers ----------------
__device__ __forceinline__ unsigned enc_ord(float f) {
    unsigned u = __float_as_uint(f);
    return (u & 0x80000000u) ? ~u : (u | 0x80000000u);
}
__device__ __forceinline__ float dec_ord(unsigned u) {
    return (u & 0x80000000u) ? __uint_as_float(u ^ 0x80000000u) : __uint_as_float(~u);
}

// ---------------- fp32 -> bf16 conversion ----------------
__global__ void k_f32_to_bf16(const float* __restrict__ in, bf16* __restrict__ out, int n) {
    int i = blockIdx.x * blockDim.x + threadIdx.x;
    if (i < n) out[i] = (bf16)in[i];
}

// ---------------- pack W[K,N] fp32 -> Wt[N,K] bf16 (transpose for WMMA B operand) ---
__global__ void k_pack_wT(const float* __restrict__ W, bf16* __restrict__ Wt, int K, int N) {
    int i = blockIdx.x * blockDim.x + threadIdx.x;
    if (i >= K * N) return;
    int k = i / N, n = i - k * N;
    Wt[(size_t)n * K + k] = (bf16)W[i];
}

// ---------------- fold W[K,256],b[256] with att[4,64] -> Mf[(K+1)*4] ----------------
// Mf[k*4+h] = sum_c W[k][64h+c]*att[h][c];  Mf[K*4+h] = sum_c b[64h+c]*att[h][c]
__global__ void k_fold_att(const float* __restrict__ W, const float* __restrict__ b,
                           const float* __restrict__ att, float* __restrict__ Mf, int K) {
    int idx = blockIdx.x * blockDim.x + threadIdx.x;
    if (idx >= (K + 1) * NH) return;
    int k = idx >> 2, h = idx & 3;
    const float* row = (k == K) ? b : (W + (size_t)k * NHC);
    const float* av  = att + h * NC;
    float s = 0.f;
#pragma unroll 4
    for (int c = 0; c < NC; ++c) s += row[h * NC + c] * av[c];
    Mf[idx] = s;
}

// ---------------- skinny projection: out[N,4] = X[N,K] @ Mf[K,4] + Mf[K] row ------
__global__ void k_proj4(const float* __restrict__ X, const float* __restrict__ Mf,
                        float* __restrict__ out, int N, int K) {
    int n = blockIdx.x * blockDim.x + threadIdx.x;
    if (n >= N) return;
    const float* xr = X + (size_t)n * K;
    float s0 = Mf[K * 4 + 0], s1 = Mf[K * 4 + 1], s2 = Mf[K * 4 + 2], s3 = Mf[K * 4 + 3];
#pragma unroll 4
    for (int k = 0; k < K; ++k) {
        float xv = xr[k];
        s0 += xv * Mf[k * 4 + 0];
        s1 += xv * Mf[k * 4 + 1];
        s2 += xv * Mf[k * 4 + 2];
        s3 += xv * Mf[k * 4 + 3];
    }
    float4 r = make_float4(s0, s1, s2, s3);
    *(float4*)(out + (size_t)n * 4) = r;
}

// ---------------- WMMA bf16 GEMM: C[M,N] = A[M,K](bf16) @ Bt[N,K](bf16)^T + bias[N]
// one wave computes a 16(M) x 64(N) tile; K stepped by 32 via v_wmma_f32_16x16x32_bf16
__global__ __launch_bounds__(256)
void k_gemm_bf16(const bf16* __restrict__ A, const bf16* __restrict__ Bt,
                 const float* __restrict__ bias, float* __restrict__ C,
                 int M, int N, int K) {
    const int lane = threadIdx.x & 31;
    const int wave = threadIdx.x >> 5;
    const int nGroups = N >> 6;
    const int totalJobs = (M >> 4) * nGroups;
    const int job = blockIdx.x * 8 + wave;
    if (job >= totalJobs) return;             // wave-uniform: EXEC stays all-ones
    const int mTile = (job / nGroups) << 4;
    const int nBase = (job % nGroups) << 6;
    const int half = lane >> 4;               // 0: K 0-15 (A and B); 1: other half
    const int l16  = lane & 15;

    const bf16* Arow = A + (size_t)(mTile + l16) * K;
    const bf16* Brow0 = Bt + (size_t)(nBase +  0 + l16) * K + half * 16;
    const bf16* Brow1 = Bt + (size_t)(nBase + 16 + l16) * K + half * 16;
    const bf16* Brow2 = Bt + (size_t)(nBase + 32 + l16) * K + half * 16;
    const bf16* Brow3 = Bt + (size_t)(nBase + 48 + l16) * K + half * 16;

    v8f acc0 = {}, acc1 = {}, acc2 = {}, acc3 = {};
#pragma unroll
    for (int k0 = 0; k0 < 256; k0 += 32) {
        union { v16bf v; v8bf h[2]; } a, b0, b1, b2, b3;
        a.h[0] = *(const v8bf*)(Arow + k0 + half * 16);
        a.h[1] = *(const v8bf*)(Arow + k0 + half * 16 + 8);
        b0.h[0] = *(const v8bf*)(Brow0 + k0); b0.h[1] = *(const v8bf*)(Brow0 + k0 + 8);
        b1.h[0] = *(const v8bf*)(Brow1 + k0); b1.h[1] = *(const v8bf*)(Brow1 + k0 + 8);
        b2.h[0] = *(const v8bf*)(Brow2 + k0); b2.h[1] = *(const v8bf*)(Brow2 + k0 + 8);
        b3.h[0] = *(const v8bf*)(Brow3 + k0); b3.h[1] = *(const v8bf*)(Brow3 + k0 + 8);
        acc0 = __builtin_amdgcn_wmma_f32_16x16x32_bf16(false, a.v, false, b0.v, (short)0, acc0, false, false);
        acc1 = __builtin_amdgcn_wmma_f32_16x16x32_bf16(false, a.v, false, b1.v, (short)0, acc1, false, false);
        acc2 = __builtin_amdgcn_wmma_f32_16x16x32_bf16(false, a.v, false, b2.v, (short)0, acc2, false, false);
        acc3 = __builtin_amdgcn_wmma_f32_16x16x32_bf16(false, a.v, false, b3.v, (short)0, acc3, false, false);
    }
    const float bv0 = bias[nBase +  0 + l16];
    const float bv1 = bias[nBase + 16 + l16];
    const float bv2 = bias[nBase + 32 + l16];
    const float bv3 = bias[nBase + 48 + l16];
    const int rBase = mTile + half * 8;
#pragma unroll
    for (int r = 0; r < 8; ++r) {
        float* Crow = C + (size_t)(rBase + r) * N + nBase + l16;
        Crow[0]  = acc0[r] + bv0;
        Crow[16] = acc1[r] + bv1;
        Crow[32] = acc2[r] + bv2;
        Crow[48] = acc3[r] + bv3;
    }
}

// ---------------- edge pass 1: alpha + segment max ----------------
__global__ void k_edge_alpha_max(const int* __restrict__ src, const int* __restrict__ dst,
                                 const float* __restrict__ a_s, const float* __restrict__ a_d,
                                 const float* __restrict__ ea, float* __restrict__ alpha_out,
                                 unsigned* __restrict__ m_u, int E) {
    int e = blockIdx.x * blockDim.x + threadIdx.x;
    if (e >= E) return;
    int s = src[e], d = dst[e];
    float4 as = *(const float4*)(a_s + (size_t)s * 4);
    float4 ad = *(const float4*)(a_d + (size_t)d * 4);
    float a0 = as.x + ad.x, a1 = as.y + ad.y, a2 = as.z + ad.z, a3 = as.w + ad.w;
    if (ea) {
        float4 ev = *(const float4*)(ea + (size_t)e * 4);
        a0 += ev.x; a1 += ev.y; a2 += ev.z; a3 += ev.w;
    }
    a0 = a0 > 0.f ? a0 : 0.2f * a0;
    a1 = a1 > 0.f ? a1 : 0.2f * a1;
    a2 = a2 > 0.f ? a2 : 0.2f * a2;
    a3 = a3 > 0.f ? a3 : 0.2f * a3;
    *(float4*)(alpha_out + (size_t)e * 4) = make_float4(a0, a1, a2, a3);
    atomicMax(m_u + (size_t)d * 4 + 0, enc_ord(a0));
    atomicMax(m_u + (size_t)d * 4 + 1, enc_ord(a1));
    atomicMax(m_u + (size_t)d * 4 + 2, enc_ord(a2));
    atomicMax(m_u + (size_t)d * 4 + 3, enc_ord(a3));
}

// ---------------- edge pass 2: exp + segment sum (ex overwrites alpha) -------------
__global__ void k_edge_exp(const int* __restrict__ dst, const unsigned* __restrict__ m_u,
                           float* __restrict__ alpha_ex, float* __restrict__ den, int E) {
    int e = blockIdx.x * blockDim.x + threadIdx.x;
    if (e >= E) return;
    int d = dst[e];
    float4 al = *(const float4*)(alpha_ex + (size_t)e * 4);
    float m0 = dec_ord(m_u[(size_t)d * 4 + 0]);
    float m1 = dec_ord(m_u[(size_t)d * 4 + 1]);
    float m2 = dec_ord(m_u[(size_t)d * 4 + 2]);
    float m3 = dec_ord(m_u[(size_t)d * 4 + 3]);
    float e0 = __expf(al.x - m0), e1 = __expf(al.y - m1);
    float e2 = __expf(al.z - m2), e3 = __expf(al.w - m3);
    *(float4*)(alpha_ex + (size_t)e * 4) = make_float4(e0, e1, e2, e3);
    atomicAdd(den + (size_t)d * 4 + 0, e0);
    atomicAdd(den + (size_t)d * 4 + 1, e1);
    atomicAdd(den + (size_t)d * 4 + 2, e2);
    atomicAdd(den + (size_t)d * 4 + 3, e3);
}

// ---------------- edge pass 3: weighted gather-scatter with head-mean folded in ----
__global__ void k_edge_aggr(const int* __restrict__ src, const int* __restrict__ dst,
                            const float* __restrict__ ex, const float* __restrict__ den,
                            const float* __restrict__ hs, float* __restrict__ aggr, int E) {
    int t = blockIdx.x * blockDim.x + threadIdx.x;
    int e = t >> 6;
    int c = t & 63;
    if (e >= E) return;
    int s = src[e], d = dst[e];
    const float* exe = ex + (size_t)e * 4;
    const float* dnd = den + (size_t)d * 4;
    float w0 = exe[0] / dnd[0], w1 = exe[1] / dnd[1];
    float w2 = exe[2] / dnd[2], w3 = exe[3] / dnd[3];
    const float* h = hs + (size_t)s * NHC;
    float v = 0.25f * (w0 * h[c] + w1 * h[64 + c] + w2 * h[128 + c] + w3 * h[192 + c]);
    atomicAdd(aggr + (size_t)d * NC + c, v);
}

// ---------------- finals ----------------
__global__ void k_add(const float* __restrict__ a, const float* __restrict__ b,
                      float* __restrict__ out, int n) {
    int i = blockIdx.x * blockDim.x + threadIdx.x;
    if (i < n) out[i] = a[i] + b[i];
}

// h_txn: M=2 tanh-attention over {out_pays, out_owns}; one wave per node
__global__ __launch_bounds__(256)
void k_final_txn(const float* __restrict__ base, const float* __restrict__ outP,
                 const float* __restrict__ outO, const float* __restrict__ W1i,
                 const float* __restrict__ b1i, const float* __restrict__ w2i,
                 float* __restrict__ hout, int N) {
    int lane = threadIdx.x & 31;
    int n = (blockIdx.x * blockDim.x + threadIdx.x) >> 5;
    if (n >= N) return;
    const float* p = outP + (size_t)n * NC;
    const float* o = outO + (size_t)n * NC;
    float sc[2];
#pragma unroll
    for (int m = 0; m < 2; ++m) {
        const float* st = m ? o : p;
        float part = 0.f;
#pragma unroll
        for (int kk = 0; kk < 2; ++kk) {
            int k = lane + kk * 32;
            float acc = b1i[k];
#pragma unroll 4
            for (int c = 0; c < NC; ++c) acc += st[c] * W1i[c * 64 + k];
            part += tanhf(acc) * w2i[k];
        }
        for (int off = 16; off; off >>= 1) part += __shfl_xor(part, off, 32);
        sc[m] = part;
    }
    float mx = fmaxf(sc[0], sc[1]);
    float e0 = __expf(sc[0] - mx), e1 = __expf(sc[1] - mx);
    float inv = 1.f / (e0 + e1);
    float w0 = e0 * inv, w1 = e1 * inv;
#pragma unroll
    for (int kk = 0; kk < 2; ++kk) {
        int c = lane + kk * 32;
        hout[(size_t)n * NC + c] = base[(size_t)n * NC + c] + w0 * p[c] + w1 * o[c];
    }
}

// ======================================================================
extern "C" void kernel_launch(void* const* d_in, const int* in_sizes, int n_in,
                              void* d_out, int out_size, void* d_ws, size_t ws_size,
                              hipStream_t stream) {
    (void)in_sizes; (void)n_in; (void)out_size; (void)ws_size;
    const float* x_user  = (const float*)d_in[0];
    const float* x_txn   = (const float*)d_in[1];
    const float* ea_pays = (const float*)d_in[2];
    const int*   ei_pays = (const int*)d_in[3];
    const int*   ei_from = (const int*)d_in[4];
    const int*   ei_owns = (const int*)d_in[5];
    const float* Ws_pays = (const float*)d_in[6];
    const float* bs_pays = (const float*)d_in[7];
    const float* Wd_pays = (const float*)d_in[8];
    const float* bd_pays = (const float*)d_in[9];
    const float* att_pays= (const float*)d_in[10];
    const float* Ws_from = (const float*)d_in[11];
    const float* bs_from = (const float*)d_in[12];
    const float* Wd_from = (const float*)d_in[13];
    const float* bd_from = (const float*)d_in[14];
    const float* att_from= (const float*)d_in[15];
    const float* Ws_owns = (const float*)d_in[16];
    const float* bs_owns = (const float*)d_in[17];
    const float* Wd_owns = (const float*)d_in[18];
    const float* bd_owns = (const float*)d_in[19];
    const float* att_owns= (const float*)d_in[20];
    const float* We_pays = (const float*)d_in[21];
    const float* be_pays = (const float*)d_in[22];
    const float* W1i     = (const float*)d_in[23];
    const float* b1i     = (const float*)d_in[24];
    const float* w2i     = (const float*)d_in[25];
    const float* Wf_user = (const float*)d_in[26];
    const float* bf_user = (const float*)d_in[27];
    const float* Wf_txn  = (const float*)d_in[28];
    const float* bf_txn  = (const float*)d_in[29];

    // ---------------- workspace bump allocator (256B aligned) ----------------
    char* ws = (char*)d_ws;
    size_t off = 0;
    auto alloc = [&](size_t bytes) -> char* {
        char* p = ws + off;
        off = (off + bytes + 255) & ~(size_t)255;
        return p;
    };
    bf16* xu_bf = (bf16*)alloc((size_t)NU_N * ND * 2);
    bf16* xt_bf = (bf16*)alloc((size_t)NT_N * ND * 2);
    bf16* WsT_pays = (bf16*)alloc((size_t)ND * NHC * 2);
    bf16* WsT_from = (bf16*)alloc((size_t)ND * NHC * 2);
    bf16* WsT_owns = (bf16*)alloc((size_t)ND * NHC * 2);
    bf16* WfT_user = (bf16*)alloc((size_t)ND * NC * 2);
    bf16* WfT_txn  = (bf16*)alloc((size_t)ND * NC * 2);
    float* Mf_s_pays = (float*)alloc((ND + 1) * 4 * 4);
    float* Mf_d_pays = (float*)alloc((ND + 1) * 4 * 4);
    float* Mf_s_from = (float*)alloc((ND + 1) * 4 * 4);
    float* Mf_d_from = (float*)alloc((ND + 1) * 4 * 4);
    float* Mf_s_owns = (float*)alloc((ND + 1) * 4 * 4);
    float* Mf_d_owns = (float*)alloc((ND + 1) * 4 * 4);
    float* Mf_e      = (float*)alloc((NDE + 1) * 4 * 4);
    float* hs_pays = (float*)alloc((size_t)NU_N * NHC * 4);
    float* hs_from = (float*)alloc((size_t)NT_N * NHC * 4);
    float* hs_owns = (float*)alloc((size_t)NU_N * NHC * 4);
    float* base_user = (float*)alloc((size_t)NU_N * NC * 4);
    float* base_txn  = (float*)alloc((size_t)NT_N * NC * 4);
    float* a_s_pays = (float*)alloc((size_t)NU_N * 4 * 4);
    float* a_d_pays = (float*)alloc((size_t)NT_N * 4 * 4);
    float* a_s_from = (float*)alloc((size_t)NT_N * 4 * 4);
    float* a_d_from = (float*)alloc((size_t)NU_N * 4 * 4);
    float* a_s_owns = (float*)alloc((size_t)NU_N * 4 * 4);
    float* a_d_owns = (float*)alloc((size_t)NT_N * 4 * 4);
    float* ea_alpha = (float*)alloc((size_t)E1_N * 4 * 4);
    float* ex_pays = (float*)alloc((size_t)E1_N * 4 * 4);
    float* ex_from = (float*)alloc((size_t)E2_N * 4 * 4);
    float* ex_owns = (float*)alloc((size_t)E3_N * 4 * 4);
    // zeroed region: [m_u x3][den x3][aggr x3]
    char* zero_begin = ws + off;
    unsigned* m_pays = (unsigned*)alloc((size_t)NT_N * 4 * 4);
    unsigned* m_from = (unsigned*)alloc((size_t)NU_N * 4 * 4);
    unsigned* m_owns = (unsigned*)alloc((size_t)NT_N * 4 * 4);
    float* den_pays = (float*)alloc((size_t)NT_N * 4 * 4);
    float* den_from = (float*)alloc((size_t)NU_N * 4 * 4);
    float* den_owns = (float*)alloc((size_t)NT_N * 4 * 4);
    float* aggr_pays = (float*)alloc((size_t)NT_N * NC * 4);
    float* aggr_from = (float*)alloc((size_t)NU_N * NC * 4);
    float* aggr_owns = (float*)alloc((size_t)NT_N * NC * 4);
    size_t zero_bytes = (size_t)((ws + off) - zero_begin);

    float* h_user = (float*)d_out;
    float* h_txn  = (float*)d_out + (size_t)NU_N * NC;

    // ---------------- stage 0: conversions / packs / folds ----------------
    k_f32_to_bf16<<<(NU_N * ND + 255) / 256, 256, 0, stream>>>(x_user, xu_bf, NU_N * ND);
    k_f32_to_bf16<<<(NT_N * ND + 255) / 256, 256, 0, stream>>>(x_txn,  xt_bf, NT_N * ND);
    k_pack_wT<<<(ND * NHC + 255) / 256, 256, 0, stream>>>(Ws_pays, WsT_pays, ND, NHC);
    k_pack_wT<<<(ND * NHC + 255) / 256, 256, 0, stream>>>(Ws_from, WsT_from, ND, NHC);
    k_pack_wT<<<(ND * NHC + 255) / 256, 256, 0, stream>>>(Ws_owns, WsT_owns, ND, NHC);
    k_pack_wT<<<(ND * NC + 255) / 256, 256, 0, stream>>>(Wf_user, WfT_user, ND, NC);
    k_pack_wT<<<(ND * NC + 255) / 256, 256, 0, stream>>>(Wf_txn,  WfT_txn,  ND, NC);
    int foldGrid = ((ND + 1) * 4 + 255) / 256;
    k_fold_att<<<foldGrid, 256, 0, stream>>>(Ws_pays, bs_pays, att_pays, Mf_s_pays, ND);
    k_fold_att<<<foldGrid, 256, 0, stream>>>(Wd_pays, bd_pays, att_pays, Mf_d_pays, ND);
    k_fold_att<<<foldGrid, 256, 0, stream>>>(Ws_from, bs_from, att_from, Mf_s_from, ND);
    k_fold_att<<<foldGrid, 256, 0, stream>>>(Wd_from, bd_from, att_from, Mf_d_from, ND);
    k_fold_att<<<foldGrid, 256, 0, stream>>>(Ws_owns, bs_owns, att_owns, Mf_s_owns, ND);
    k_fold_att<<<foldGrid, 256, 0, stream>>>(Wd_owns, bd_owns, att_owns, Mf_d_owns, ND);
    k_fold_att<<<((NDE + 1) * 4 + 255) / 256, 256, 0, stream>>>(We_pays, be_pays, att_pays, Mf_e, NDE);

    // ---------------- stage 1: WMMA GEMMs ----------------
    auto gemmBlocks = [](int M, int N) { return ((M >> 4) * (N >> 6) + 7) / 8; };
    k_gemm_bf16<<<gemmBlocks(NU_N, NHC), 256, 0, stream>>>(xu_bf, WsT_pays, bs_pays, hs_pays, NU_N, NHC, ND);
    k_gemm_bf16<<<gemmBlocks(NT_N, NHC), 256, 0, stream>>>(xt_bf, WsT_from, bs_from, hs_from, NT_N, NHC, ND);
    k_gemm_bf16<<<gemmBlocks(NU_N, NHC), 256, 0, stream>>>(xu_bf, WsT_owns, bs_owns, hs_owns, NU_N, NHC, ND);
    k_gemm_bf16<<<gemmBlocks(NU_N, NC), 256, 0, stream>>>(xu_bf, WfT_user, bf_user, base_user, NU_N, NC, ND);
    k_gemm_bf16<<<gemmBlocks(NT_N, NC), 256, 0, stream>>>(xt_bf, WfT_txn,  bf_txn,  base_txn,  NT_N, NC, ND);

    // ---------------- stage 2: attention logits (folded skinny projections) ----------
    k_proj4<<<(NU_N + 255) / 256, 256, 0, stream>>>(x_user, Mf_s_pays, a_s_pays, NU_N, ND);
    k_proj4<<<(NT_N + 255) / 256, 256, 0, stream>>>(x_txn,  Mf_d_pays, a_d_pays, NT_N, ND);
    k_proj4<<<(NT_N + 255) / 256, 256, 0, stream>>>(x_txn,  Mf_s_from, a_s_from, NT_N, ND);
    k_proj4<<<(NU_N + 255) / 256, 256, 0, stream>>>(x_user, Mf_d_from, a_d_from, NU_N, ND);
    k_proj4<<<(NU_N + 255) / 256, 256, 0, stream>>>(x_user, Mf_s_owns, a_s_owns, NU_N, ND);
    k_proj4<<<(NT_N + 255) / 256, 256, 0, stream>>>(x_txn,  Mf_d_owns, a_d_owns, NT_N, ND);
    k_proj4<<<(E1_N + 255) / 256, 256, 0, stream>>>(ea_pays, Mf_e, ea_alpha, E1_N, NDE);

    // ---------------- stage 3: segment softmax + aggregation ----------------
    hipMemsetAsync(zero_begin, 0, zero_bytes, stream);
    const int* srcP = ei_pays;          const int* dstP = ei_pays + E1_N;  // user -> txn
    const int* srcF = ei_from;          const int* dstF = ei_from + E2_N;  // txn  -> user
    const int* srcO = ei_owns;          const int* dstO = ei_owns + E3_N;  // user -> txn
    int eg = (E1_N + 255) / 256;
    k_edge_alpha_max<<<eg, 256, 0, stream>>>(srcP, dstP, a_s_pays, a_d_pays, ea_alpha, ex_pays, m_pays, E1_N);
    k_edge_alpha_max<<<eg, 256, 0, stream>>>(srcF, dstF, a_s_from, a_d_from, nullptr,  ex_from, m_from, E2_N);
    k_edge_alpha_max<<<eg, 256, 0, stream>>>(srcO, dstO, a_s_owns, a_d_owns, nullptr,  ex_owns, m_owns, E3_N);
    k_edge_exp<<<eg, 256, 0, stream>>>(dstP, m_pays, ex_pays, den_pays, E1_N);
    k_edge_exp<<<eg, 256, 0, stream>>>(dstF, m_from, ex_from, den_from, E2_N);
    k_edge_exp<<<eg, 256, 0, stream>>>(dstO, m_owns, ex_owns, den_owns, E3_N);
    int ag = ((size_t)E1_N * 64 + 255) / 256;
    k_edge_aggr<<<ag, 256, 0, stream>>>(srcP, dstP, ex_pays, den_pays, hs_pays, aggr_pays, E1_N);
    k_edge_aggr<<<ag, 256, 0, stream>>>(srcF, dstF, ex_from, den_from, hs_from, aggr_from, E2_N);
    k_edge_aggr<<<ag, 256, 0, stream>>>(srcO, dstO, ex_owns, den_owns, hs_owns, aggr_owns, E3_N);

    // ---------------- stage 4: finals ----------------
    // M=1 inter-attention degenerates to identity weight -> simple add
    k_add<<<(NU_N * NC + 255) / 256, 256, 0, stream>>>(base_user, aggr_from, h_user, NU_N * NC);
    k_final_txn<<<(NT_N * 32 + 255) / 256, 256, 0, stream>>>(base_txn, aggr_pays, aggr_owns,
                                                             W1i, b1i, w2i, h_txn, NT_N);
}
